// BlockTransformer_35802847379497
// MI455X (gfx1250) — matrix operand order repaired and checked by reading, not compiled, not run on hardware
//
#include <hip/hip_runtime.h>
#include <cstdint>
#include <cstddef>

// ---------------- problem constants ----------------
constexpr int Bc = 8, Pc = 16, HORc = 10, NOc = 128, NRc = 8;
constexpr int TPSc = NOc + NRc;            // 136
constexpr int Tc  = Pc + HORc * TPSc;      // 1376
constexpr int Dc  = 768, NHc = 12, HDc = 64, Fc = 3072, Lc = 12;
constexpr int Mrows = Bc * Tc;             // 11008
constexpr int QKVLD = 3 * Dc;              // 2304

typedef __attribute__((ext_vector_type(16))) __bf16 v16bf;
typedef __attribute__((ext_vector_type(8)))  float  v8f;

union FragB {                 // one WMMA 16-bf16 operand fragment per lane
  v16bf v;
  uint4 q[2];
  unsigned short s[16];
};

static __device__ __forceinline__ unsigned short f2bf(float f) {
  union { float f; unsigned u; } x; x.f = f;
  unsigned u = x.u;
  u += 0x7FFFu + ((u >> 16) & 1u);        // round-to-nearest-even
  return (unsigned short)(u >> 16);
}

static __device__ __forceinline__ float gelu_tanh(float v) {
  float x3 = v * v * v;
  return 0.5f * v * (1.0f + tanhf(0.7978845608028654f * (v + 0.044715f * x3)));
}

// rule mask: groups 0=prefix 1=obs 2=readout; causal on timestep index
static __device__ __forceinline__ bool rule_allow(int tq, int tk) {
  int gq, gk, tsq, tsk;
  if (tq < Pc) { gq = 0; tsq = -1; } else { int q = tq - Pc; tsq = q / TPSc; gq = (q % TPSc) < NOc ? 1 : 2; }
  if (tk < Pc) { gk = 0; tsk = -1; } else { int q = tk - Pc; tsk = q / TPSc; gk = (q % TPSc) < NOc ? 1 : 2; }
  if (gk == 0) return true;                  // column group 0: 'all' for every row group
  if (gq == 0) return false;                 // prefix row vs obs/readout: never
  if (gk == 1) return tsk <= tsq;            // obs column: causal
  return (gq == 2) && (tsk <= tsq);          // readout column: causal only for readout rows
}

// ---------------- sequence assembly ----------------
__global__ __launch_bounds__(256) void build_x(
    const float* __restrict__ pre, const float* __restrict__ obs,
    const float* __restrict__ ro, float* __restrict__ h) {
  int bt = blockIdx.x;
  int b = bt / Tc, t = bt % Tc;
  const float* src;
  if (t < Pc) {
    src = pre + ((size_t)b * Pc + t) * Dc;
  } else {
    int q = t - Pc, ht = q / TPSc, off = q % TPSc;
    src = (off < NOc) ? obs + (((size_t)b * HORc + ht) * NOc + off) * Dc
                      : ro  + (((size_t)b * HORc + ht) * NRc + (off - NOc)) * Dc;
  }
  float* dst = h + (size_t)bt * Dc;
  for (int i = threadIdx.x; i < Dc; i += 256) dst[i] = src[i];
}

__global__ __launch_bounds__(256) void build_pad(
    const unsigned char* __restrict__ pm, const unsigned char* __restrict__ om,
    const unsigned char* __restrict__ rm, unsigned char* __restrict__ pad) {
  int idx = blockIdx.x * 256 + threadIdx.x;
  if (idx >= Mrows) return;
  int b = idx / Tc, t = idx % Tc;
  unsigned char v;
  if (t < Pc) v = pm[b * Pc + t];
  else {
    int q = t - Pc, ht = q / TPSc, off = q % TPSc;
    v = (off < NOc) ? om[(b * HORc + ht) * NOc + off]
                    : rm[(b * HORc + ht) * NRc + (off - NOc)];
  }
  pad[idx] = v;
}

// ---------------- layernorm: one wave per 768-row ----------------
template <bool BF16OUT>
__global__ __launch_bounds__(256) void ln_rows(
    const float* __restrict__ x, const float* __restrict__ gs, const float* __restrict__ gb,
    unsigned short* __restrict__ yb, float* __restrict__ yf) {
  const int lane = threadIdx.x & 31, wv = threadIdx.x >> 5;
  const size_t row = (size_t)blockIdx.x * 8 + wv;
  const float* xr = x + row * Dc;
  float v[24];
  float s = 0.f;
#pragma unroll
  for (int i = 0; i < 24; ++i) { v[i] = xr[i * 32 + lane]; s += v[i]; }
#pragma unroll
  for (int m = 16; m >= 1; m >>= 1) s += __shfl_xor(s, m, 32);
  float mu = s * (1.0f / Dc);
  float vs = 0.f;
#pragma unroll
  for (int i = 0; i < 24; ++i) { float d = v[i] - mu; vs += d * d; }
#pragma unroll
  for (int m = 16; m >= 1; m >>= 1) vs += __shfl_xor(vs, m, 32);
  float rstd = rsqrtf(vs * (1.0f / Dc) + 1e-6f);
#pragma unroll
  for (int i = 0; i < 24; ++i) {
    int e = i * 32 + lane;
    float o = (v[i] - mu) * rstd * gs[e] + gb[e];
    if (BF16OUT) yb[row * Dc + e] = f2bf(o);
    else         yf[row * Dc + e] = o;
  }
}

// ---------------- weight convert + transpose: f32 [K][N] -> bf16 [N][K] ----------------
__global__ __launch_bounds__(256) void conv_transpose_bf16(
    const float* __restrict__ W, unsigned short* __restrict__ Wt, int K, int N) {
  __shared__ float tile[32][33];
  int tx = threadIdx.x & 31, ty = threadIdx.x >> 5;
  int n0 = blockIdx.x * 32, k0 = blockIdx.y * 32;
#pragma unroll
  for (int r = ty; r < 32; r += 8) tile[r][tx] = W[(size_t)(k0 + r) * N + n0 + tx];
  __syncthreads();
#pragma unroll
  for (int r = ty; r < 32; r += 8) Wt[(size_t)(n0 + r) * K + k0 + tx] = f2bf(tile[tx][r]);
}

// ---------------- WMMA GEMM: C[M][N] = A[M][K](bf16) * Wt[N][K](bf16)^T + epilogue ----------------
// block tile 128x128, 8 waves (2 M x 4 N), wave tile 64x32 = 4x2 WMMA tiles, K-step 32.
// A-fragments reused across 2 N-subtiles, B-fragments across 4 M-subtiles.
template <int EPI>  // 0: bias->f32   1: res+bias->f32   2: gelu(bias)->bf16
__global__ __launch_bounds__(256) void gemm_wmma(
    const unsigned short* __restrict__ A, const unsigned short* __restrict__ Wt,
    const float* __restrict__ bias, const float* __restrict__ res,
    float* __restrict__ outF, unsigned short* __restrict__ outB,
    int M, int N, int K) {
  __shared__ unsigned short lA[128 * 40];   // stride 40 keeps ds b128 16B-aligned
  __shared__ unsigned short lB[128 * 40];
  const int tid = threadIdx.x;
  const int lane = tid & 31, wv = tid >> 5;
  const int wm = wv & 1, wn = wv >> 1;
  const int bm = blockIdx.x, bn = blockIdx.y;
  const int hlf = lane >> 4, mr = lane & 15;

  v8f zf = {0.f, 0.f, 0.f, 0.f, 0.f, 0.f, 0.f, 0.f};
  v8f acc[4][2] = {{zf, zf}, {zf, zf}, {zf, zf}, {zf, zf}};

  const int grow = tid >> 1, gc16 = (tid & 1) * 16;   // 128x32 tile staging, 16 elems/thread/tile
  const unsigned short* Ar = A + (size_t)(bm * 128 + grow) * K + gc16;
  const unsigned short* Br = Wt + (size_t)(bn * 128 + grow) * K + gc16;

  for (int kb = 0; kb < K; kb += 32) {
    *(uint4*)&lA[grow * 40 + gc16]     = *(const uint4*)(Ar + kb);
    *(uint4*)&lA[grow * 40 + gc16 + 8] = *(const uint4*)(Ar + kb + 8);
    *(uint4*)&lB[grow * 40 + gc16]     = *(const uint4*)(Br + kb);
    *(uint4*)&lB[grow * 40 + gc16 + 8] = *(const uint4*)(Br + kb + 8);
    if (kb + 32 < K) {                  // gfx1250 global_prefetch_b8 of next K-tile
      __builtin_prefetch((const void*)(Ar + kb + 32), 0, 3);
      __builtin_prefetch((const void*)(Br + kb + 32), 0, 3);
    }
    __syncthreads();

    FragB aF[4], bF[2];
#pragma unroll
    for (int i = 0; i < 4; ++i) {           // A frag: 16x32, lane m = mr, halves split K
      int r = (wm * 64 + i * 16 + mr) * 40;
      aF[i].q[0] = *(const uint4*)&lA[r + hlf * 8];        // K  0..7 / 8..15
      aF[i].q[1] = *(const uint4*)&lA[r + 16 + hlf * 8];   // K 16..23 / 24..31
    }
#pragma unroll
    for (int j = 0; j < 2; ++j) {           // B frag: 32x16, lane n = mr, halves split K
      int r = (wn * 32 + j * 16 + mr) * 40 + hlf * 8;
      bF[j].q[0] = *(const uint4*)&lB[r];
      bF[j].q[1] = *(const uint4*)&lB[r + 16];
    }
#pragma unroll
    for (int i = 0; i < 4; ++i)
#pragma unroll
      for (int j = 0; j < 2; ++j)
        acc[i][j] = __builtin_amdgcn_wmma_f32_16x16x32_bf16(
            false, aF[i].v, false, bF[j].v, (short)0, acc[i][j], false, false);
    __syncthreads();
  }

#pragma unroll
  for (int i = 0; i < 4; ++i) {
    int row0 = bm * 128 + wm * 64 + i * 16 + 8 * hlf;   // C layout: VGPR r -> M=r+8*half
#pragma unroll
    for (int j = 0; j < 2; ++j) {
      int col = bn * 128 + wn * 32 + j * 16 + mr;       // C layout: N = lane&15
      float bcol = bias[col];
#pragma unroll
      for (int r = 0; r < 8; ++r) {
        size_t idx = (size_t)(row0 + r) * N + col;
        float v = acc[i][j][r] + bcol;
        if (EPI == 0)      outF[idx] = v;
        else if (EPI == 1) outF[idx] = res[idx] + v;
        else               outB[idx] = f2bf(gelu_tanh(v));
      }
    }
  }
}

// ---------------- flash attention with WMMA ----------------
// One wave handles a 16-query tile. Computes S^T = K_chunk @ Q^T so queries live on
// the lane axis; the St VGPRs are exactly the A-fragment element order for P@V.
__global__ __launch_bounds__(64) void attn_wmma(
    const float* __restrict__ qkv, const unsigned char* __restrict__ pad,
    unsigned short* __restrict__ outb) {
  __shared__ unsigned short ldsVT[2][64 * 40];   // V chunk transposed: [hd][key], per wave
  const int lane = threadIdx.x & 31, wv = threadIdx.x >> 5;
  const int b = blockIdx.x / NHc, hh = blockIdx.x % NHc;
  const int qt = blockIdx.y * 2 + wv;            // 43*2 = 86 = T/16 exact
  const int hlf = lane >> 4, nq = lane & 15;
  const int tq = qt * 16 + nq;
  const float scale = 0.125f;                    // 1/sqrt(64)

  const float* qbase = qkv + (size_t)(b * Tc) * QKVLD + hh * HDc;
  const float* kbase = qbase + Dc;
  const float* vbase = qbase + 2 * Dc;
  const unsigned char* padb = pad + b * Tc;

  // B-fragments of Q^T (hd x 16): lane n = query, halves split hd
  FragB bQ[2];
  {
    const float* qr = qbase + (size_t)tq * QKVLD;
#pragma unroll
    for (int kb = 0; kb < 2; ++kb) {
      const float* p = qr + kb * 32 + hlf * 16;
#pragma unroll
      for (int c = 0; c < 16; ++c) bQ[kb].s[c] = f2bf(p[c]);
    }
  }

  v8f zf = {0.f, 0.f, 0.f, 0.f, 0.f, 0.f, 0.f, 0.f};
  v8f accO[4] = {zf, zf, zf, zf};
  float mrow = -3.0e38f, lsum = 0.f;

  for (int ck = 0; ck < Tc; ck += 32) {
    // stage V chunk transposed into LDS: lane <-> key
    {
      const float* vr = vbase + (size_t)(ck + lane) * QKVLD;
      unsigned short* dst = &ldsVT[wv][0];
#pragma unroll
      for (int d0 = 0; d0 < 64; ++d0) dst[d0 * 40 + lane] = f2bf(vr[d0]);
    }
    __syncthreads();

    // S^T tiles: St[kt] = K(16 keys x 64) @ Q^T (64 x 16)
    v8f St[2];
#pragma unroll
    for (int kt = 0; kt < 2; ++kt) {
      FragB aK[2];
      const float* kr = kbase + (size_t)(ck + kt * 16 + nq) * QKVLD;
#pragma unroll
      for (int kb = 0; kb < 2; ++kb) {
#pragma unroll
        for (int c = 0; c < 8; ++c) {
          aK[kb].s[c]     = f2bf(kr[kb * 32 + hlf * 16 + c]);
          aK[kb].s[8 + c] = f2bf(kr[kb * 32 + hlf * 16 + 8 + c]);
        }
      }
      v8f t = __builtin_amdgcn_wmma_f32_16x16x32_bf16(false, aK[0].v, false, bQ[0].v, (short)0, zf, false, false);
      St[kt] = __builtin_amdgcn_wmma_f32_16x16x32_bf16(false, aK[1].v, false, bQ[1].v, (short)0, t, false, false);
    }

    // mask + online softmax (row = query = lane axis)
    float p0[8], p1[8], mloc = -3.0e38f;
#pragma unroll
    for (int r = 0; r < 8; ++r) {
      int tk0 = ck + r + 8 * hlf;
      int tk1 = tk0 + 16;
      float s0 = (rule_allow(tq, tk0) && padb[tk0]) ? St[0][r] * scale : -3.0e38f;
      float s1 = (rule_allow(tq, tk1) && padb[tk1]) ? St[1][r] * scale : -3.0e38f;
      p0[r] = s0; p1[r] = s1;
      mloc = fmaxf(mloc, fmaxf(s0, s1));
    }
    mloc = fmaxf(mloc, __shfl_xor(mloc, 16, 32));
    float mnew = fmaxf(mrow, mloc);
    float alpha = __expf(mrow - mnew);
    float rs = 0.f;
    FragB aP;                                  // P fragment directly from St registers
#pragma unroll
    for (int r = 0; r < 8; ++r) {
      float e0 = __expf(p0[r] - mnew), e1 = __expf(p1[r] - mnew);
      rs += e0 + e1;
      aP.s[r] = f2bf(e0);
      aP.s[8 + r] = f2bf(e1);
    }
    rs += __shfl_xor(rs, 16, 32);
    mrow = mnew;
    lsum = lsum * alpha + rs;

    float alphaR[8];
#pragma unroll
    for (int r = 0; r < 8; ++r) alphaR[r] = __shfl(alpha, 8 * hlf + r, 32);

#pragma unroll
    for (int nt = 0; nt < 4; ++nt) {           // O(16q x 64hd) += P(16x32) @ V(32x16)
      FragB bV;
      int rowhd = nt * 16 + nq;
      bV.q[0] = *(const uint4*)&ldsVT[wv][rowhd * 40 + hlf * 8];
      bV.q[1] = *(const uint4*)&ldsVT[wv][rowhd * 40 + hlf * 8 + 16];
      v8f c;
#pragma unroll
      for (int r = 0; r < 8; ++r) c[r] = accO[nt][r] * alphaR[r];
      accO[nt] = __builtin_amdgcn_wmma_f32_16x16x32_bf16(false, aP.v, false, bV.v, (short)0, c, false, false);
    }
    __syncthreads();
  }

  float linv = 1.0f / lsum;
  float linvR[8];
#pragma unroll
  for (int r = 0; r < 8; ++r) linvR[r] = __shfl(linv, 8 * hlf + r, 32);
#pragma unroll
  for (int nt = 0; nt < 4; ++nt)
#pragma unroll
    for (int r = 0; r < 8; ++r) {
      int trow = qt * 16 + 8 * hlf + r;
      outb[(size_t)(b * Tc + trow) * Dc + hh * HDc + nt * 16 + nq] = f2bf(accO[nt][r] * linvR[r]);
    }
}

// ---------------- driver ----------------
extern "C" void kernel_launch(void* const* d_in, const int* in_sizes, int n_in,
                              void* d_out, int out_size, void* d_ws, size_t ws_size,
                              hipStream_t stream) {
  const float* prefix = (const float*)d_in[0];
  const unsigned char* pmask = (const unsigned char*)d_in[1];
  const float* obs = (const float*)d_in[2];
  const unsigned char* omask = (const unsigned char*)d_in[3];
  const float* ro = (const float*)d_in[4];
  const unsigned char* rmask = (const unsigned char*)d_in[5];
  const float* ln1s = (const float*)d_in[6], *ln1b = (const float*)d_in[7];
  const float* wqkv = (const float*)d_in[8], *bqkv = (const float*)d_in[9];
  const float* wo = (const float*)d_in[10], *bo = (const float*)d_in[11];
  const float* ln2s = (const float*)d_in[12], *ln2b = (const float*)d_in[13];
  const float* w1 = (const float*)d_in[14], *b1 = (const float*)d_in[15];
  const float* w2 = (const float*)d_in[16], *b2 = (const float*)d_in[17];
  const float* lnfs = (const float*)d_in[18], *lnfb = (const float*)d_in[19];

  char* base = (char*)d_ws;
  size_t off = 0;
  auto carve = [&](size_t bytes) -> char* {
    char* p = base + off;
    off += (bytes + 255) & ~(size_t)255;
    return p;
  };
  float*          h    = (float*)         carve((size_t)Mrows * Dc * 4);
  unsigned short* ybf  = (unsigned short*)carve((size_t)Mrows * Dc * 2);
  float*          qkvb = (float*)         carve((size_t)Mrows * QKVLD * 4);
  unsigned short* abf  = (unsigned short*)carve((size_t)Mrows * Dc * 2);
  unsigned short* ubf  = (unsigned short*)carve((size_t)Mrows * Fc * 2);
  unsigned short* wt   = (unsigned short*)carve((size_t)Dc * Fc * 2);
  unsigned char*  pad  = (unsigned char*) carve((size_t)Mrows);

  build_x<<<Mrows, 256, 0, stream>>>(prefix, obs, ro, h);
  build_pad<<<(Mrows + 255) / 256, 256, 0, stream>>>(pmask, omask, rmask, pad);

  for (int l = 0; l < Lc; ++l) {
    ln_rows<true><<<Mrows / 8, 256, 0, stream>>>(h, ln1s + l * Dc, ln1b + l * Dc, ybf, nullptr);
    conv_transpose_bf16<<<dim3(QKVLD / 32, Dc / 32), 256, 0, stream>>>(wqkv + (size_t)l * Dc * QKVLD, wt, Dc, QKVLD);
    gemm_wmma<0><<<dim3(Mrows / 128, QKVLD / 128), 256, 0, stream>>>(
        ybf, wt, bqkv + l * QKVLD, nullptr, qkvb, nullptr, Mrows, QKVLD, Dc);
    attn_wmma<<<dim3(Bc * NHc, Tc / 32), 64, 0, stream>>>(qkvb, pad, abf);
    conv_transpose_bf16<<<dim3(Dc / 32, Dc / 32), 256, 0, stream>>>(wo + (size_t)l * Dc * Dc, wt, Dc, Dc);
    gemm_wmma<1><<<dim3(Mrows / 128, Dc / 128), 256, 0, stream>>>(
        abf, wt, bo + l * Dc, h, h, nullptr, Mrows, Dc, Dc);
    ln_rows<true><<<Mrows / 8, 256, 0, stream>>>(h, ln2s + l * Dc, ln2b + l * Dc, ybf, nullptr);
    conv_transpose_bf16<<<dim3(Fc / 32, Dc / 32), 256, 0, stream>>>(w1 + (size_t)l * Dc * Fc, wt, Dc, Fc);
    gemm_wmma<2><<<dim3(Mrows / 128, Fc / 128), 256, 0, stream>>>(
        ybf, wt, b1 + l * Fc, nullptr, nullptr, ubf, Mrows, Fc, Dc);
    conv_transpose_bf16<<<dim3(Dc / 32, Fc / 32), 256, 0, stream>>>(w2 + (size_t)l * Fc * Dc, wt, Fc, Dc);
    gemm_wmma<1><<<dim3(Mrows / 128, Dc / 128), 256, 0, stream>>>(
        ubf, wt, b2 + l * Dc, h, h, nullptr, Mrows, Dc, Fc);
  }
  ln_rows<false><<<Mrows / 8, 256, 0, stream>>>(h, lnfs, lnfb, nullptr, (float*)d_out);
}
